// BEVFormerLayer_5179730559500
// MI455X (gfx1250) — matrix operand order repaired and checked by reading, not compiled, not run on hardware
//
#include <hip/hip_runtime.h>
#include <hip/hip_bf16.h>
#include <math.h>

// ---------------------------------------------------------------------------
// BEVFormer encoder layer for MI455X (gfx1250, wave32, WMMA).
// GEMMs: v_wmma_f32_16x16x32_f16, f32 accumulate. Weights pre-converted to
// f16; per-block 64xK f16 B-panel staged in LDS via async global->LDS
// (ASYNCcnt) so A global loads / B ds loads / WMMA overlap on separate
// counters. Deformable gathers stay scalar: the 120MB projected value tensor
// is L2-resident (192MB L2).
// ---------------------------------------------------------------------------

typedef __attribute__((ext_vector_type(16))) _Float16 v16h;
typedef __attribute__((ext_vector_type(8)))  _Float16 v8h;
typedef __attribute__((ext_vector_type(8)))  float    v8f;
typedef __attribute__((ext_vector_type(4)))  float    v4f;
typedef int v4i __attribute__((vector_size(16)));   // matches async-LDS builtin proto

#define NQ      10000
#define C_DIM   256
#define HEADS   8
#define HDIM    32
#define CAMS    6
#define LEVELS  4
#define NUM_VAL 19560
#define TPTS    4
#define SPTS    8
#define FF_DIM  512

#if defined(__has_builtin)
#  if __has_builtin(__builtin_amdgcn_global_load_async_to_lds_b128) && \
      __has_builtin(__builtin_amdgcn_s_wait_asynccnt)
#    define USE_ASYNC_LDS 1
#  endif
#  if __has_builtin(__builtin_amdgcn_sched_barrier)
#    define SCHED_FENCE() __builtin_amdgcn_sched_barrier(0)
#  endif
#endif
#ifndef USE_ASYNC_LDS
#  define USE_ASYNC_LDS 0
#endif
#ifndef SCHED_FENCE
#  define SCHED_FENCE() ((void)0)
#endif

__device__ __constant__ int kLvlH[LEVELS] = {92, 46, 23, 12};
__device__ __constant__ int kLvlW[LEVELS] = {160, 80, 40, 20};
__device__ __constant__ int kLvlS[LEVELS] = {0, 14720, 18400, 19320};

// ---------------------------------------------------------------------------
// Weight f32 -> f16 pre-conversion.
// ---------------------------------------------------------------------------
__global__ void cvt_f16_kernel(const float* __restrict__ s,
                               _Float16* __restrict__ d, int n)
{
  int i = blockIdx.x * blockDim.x + threadIdx.x;
  if (i < n) d[i] = (_Float16)s[i];
}

// ---------------------------------------------------------------------------
// WMMA GEMM:  out[M,N] = A[M,K] @ Wh[N,K]^T + bias[N]   (act=1 -> ReLU)
// A: f32 (converted on the fly). Wh: f16. Block: 256 thr = 8 waves; block
// owns 64 output columns; the 64xK f16 weight panel is async-staged in LDS.
// Requires: M % 16 == 0, N % 64 == 0, K % 32 == 0 (all callsites satisfy).
// Dynamic LDS: 64*K*2 bytes.
// ---------------------------------------------------------------------------
__global__ __launch_bounds__(256) void gemm_wmma_kernel(
    const float* __restrict__ A, const _Float16* __restrict__ Wh,
    const float* __restrict__ bias, float* __restrict__ out,
    int M, int N, int K, int act)
{
  extern __shared__ _Float16 bpanel[];   // 64 rows (n) x K halves

  const int n0 = blockIdx.x * 64;

  // ---- Stage the 64xK f16 weight panel into LDS (rows n0..n0+63 are
  //      contiguous in Wh since it is row-major [N,K]).
  {
    const _Float16* gpanel = Wh + (size_t)n0 * K;
    const int nchunk = (64 * K) / 8;     // 16-byte chunks
#if USE_ASYNC_LDS
    for (int i = threadIdx.x; i < nchunk; i += 256) {
      __builtin_amdgcn_global_load_async_to_lds_b128(
          (__attribute__((address_space(1))) v4i*)(gpanel + (size_t)i * 8),
          (__attribute__((address_space(3))) v4i*)(&bpanel[(size_t)i * 8]),
          0, 0);
    }
    __builtin_amdgcn_s_wait_asynccnt(0);
#else
    for (int i = threadIdx.x; i < nchunk; i += 256) {
      *(v8h*)(&bpanel[(size_t)i * 8]) = *(const v8h*)(gpanel + (size_t)i * 8);
    }
#endif
  }
  __syncthreads();

  const int wave = threadIdx.x >> 5;
  const int lane = threadIdx.x & 31;
  const int half = lane >> 4;            // 0: lanes 0-15, 1: lanes 16-31
  const int r    = lane & 15;

  const int m0 = (blockIdx.y * 8 + wave) * 16;
  if (m0 >= M) return;

  v8f acc[4] = {};
  const float* arow = A + (size_t)(m0 + r) * K;

#pragma unroll 2
  for (int k0 = 0; k0 < K; k0 += 32) {
    if (k0 + 32 < K) __builtin_prefetch(arow + k0 + 32, 0, 3);

    // ---- A fragment (16x32 f16): lanes 0-15 hold K {0..15},
    //      lanes 16-31 hold K {16..31} of this k-chunk.
    v4f a0 = *(const v4f*)(arow + k0 + half * 16);
    v4f a1 = *(const v4f*)(arow + k0 + half * 16 + 4);
    v4f a2 = *(const v4f*)(arow + k0 + half * 16 + 8);
    v4f a3 = *(const v4f*)(arow + k0 + half * 16 + 12);

    // ---- Issue ALL B fragment ds reads first (distinct regs -> one clause,
    //      staggered dscnt waits; fills WMMA hazard slots).
    const _Float16* bbase = bpanel + k0 + half * 16;
    v16h bf0 = *(const v16h*)(bbase + (size_t)(0 * 16 + r) * K);
    v16h bf1 = *(const v16h*)(bbase + (size_t)(1 * 16 + r) * K);
    v16h bf2 = *(const v16h*)(bbase + (size_t)(2 * 16 + r) * K);
    v16h bf3 = *(const v16h*)(bbase + (size_t)(3 * 16 + r) * K);

    v16h afrag;
#pragma unroll
    for (int i = 0; i < 4; ++i) {
      afrag[i]      = (_Float16)a0[i];
      afrag[4 + i]  = (_Float16)a1[i];
      afrag[8 + i]  = (_Float16)a2[i];
      afrag[12 + i] = (_Float16)a3[i];
    }

    // Keep the ds_loads above from being sunk in between the WMMAs: all 8
    // ds_load_b128 issue up front, one dscnt exposure per k-step.
    SCHED_FENCE();

    acc[0] = __builtin_amdgcn_wmma_f32_16x16x32_f16(
        false, afrag, false, bf0, (short)0, acc[0], false, false);
    acc[1] = __builtin_amdgcn_wmma_f32_16x16x32_f16(
        false, afrag, false, bf1, (short)0, acc[1], false, false);
    acc[2] = __builtin_amdgcn_wmma_f32_16x16x32_f16(
        false, afrag, false, bf2, (short)0, acc[2], false, false);
    acc[3] = __builtin_amdgcn_wmma_f32_16x16x32_f16(
        false, afrag, false, bf3, (short)0, acc[3], false, false);
  }

  // ---- Store. C/D layout: VGPR rr, lanes 0-15 -> M=rr, lanes 16-31 -> M=8+rr.
#pragma unroll
  for (int t = 0; t < 4; ++t) {
    const int n  = n0 + t * 16 + r;
    const float bn = bias ? bias[n] : 0.0f;
#pragma unroll
    for (int rr = 0; rr < 8; ++rr) {
      const int m = m0 + half * 8 + rr;
      float v = acc[t][rr] + bn;
      if (act == 1) v = v > 0.0f ? v : 0.0f;
      out[(size_t)m * N + n] = v;
    }
  }
}

// ---------------------------------------------------------------------------
// qc = concat([query, query + bev_pos], -1)   -> (NQ, 512)
// ---------------------------------------------------------------------------
__global__ void build_qc_kernel(const float* __restrict__ query,
                                const float* __restrict__ bev_pos,
                                float* __restrict__ qc)
{
  int idx = blockIdx.x * blockDim.x + threadIdx.x;
  if (idx >= NQ * 2 * C_DIM) return;
  int q = idx / (2 * C_DIM);
  int j = idx % (2 * C_DIM);
  float v;
  if (j < C_DIM) v = query[q * C_DIM + j];
  else {
    int jj = j - C_DIM;
    v = query[q * C_DIM + jj] + bev_pos[q * C_DIM + jj];
  }
  qc[idx] = v;
}

// ---------------------------------------------------------------------------
// In-place softmax over contiguous groups of size gsz.
// ---------------------------------------------------------------------------
__global__ void softmax_groups_kernel(float* __restrict__ x, int ngroups, int gsz)
{
  int g = blockIdx.x * blockDim.x + threadIdx.x;
  if (g >= ngroups) return;
  float* p = x + (size_t)g * gsz;
  float mx = p[0];
  for (int i = 1; i < gsz; ++i) mx = fmaxf(mx, p[i]);
  float s = 0.0f;
  for (int i = 0; i < gsz; ++i) { float e = expf(p[i] - mx); p[i] = e; s += e; }
  float inv = 1.0f / s;
  for (int i = 0; i < gsz; ++i) p[i] *= inv;
}

// ---------------------------------------------------------------------------
// Fused (x + res) -> LayerNorm(gamma, beta). One block (256 thr) per row.
// Wave32 shuffle reduction + LDS across 8 waves.
// ---------------------------------------------------------------------------
__global__ __launch_bounds__(256) void add_ln_kernel(
    const float* __restrict__ x, const float* __restrict__ res,
    const float* __restrict__ g, const float* __restrict__ b,
    float* __restrict__ out)
{
  __shared__ float r1[8], r2[8];
  const int q = blockIdx.x;
  const int t = threadIdx.x;
  float v = x[(size_t)q * C_DIM + t];
  if (res) v += res[(size_t)q * C_DIM + t];
  float s1 = v, s2 = v * v;
#pragma unroll
  for (int o = 16; o > 0; o >>= 1) {
    s1 += __shfl_xor(s1, o, 32);
    s2 += __shfl_xor(s2, o, 32);
  }
  if ((t & 31) == 0) { r1[t >> 5] = s1; r2[t >> 5] = s2; }
  __syncthreads();
  float S1 = 0.0f, S2 = 0.0f;
#pragma unroll
  for (int i = 0; i < 8; ++i) { S1 += r1[i]; S2 += r2[i]; }
  const float mean = S1 * (1.0f / C_DIM);
  const float var  = S2 * (1.0f / C_DIM) - mean * mean;
  const float inv  = rsqrtf(var + 1e-5f);
  out[(size_t)q * C_DIM + t] = (v - mean) * inv * g[t] + b[t];
}

// ---------------------------------------------------------------------------
// TSA deformable attention: 2 temporal branches over the 100x100 BEV grid,
// averaged. Block per query, thread = (head h = tid/32, dim d = tid%32).
// ---------------------------------------------------------------------------
__global__ __launch_bounds__(256) void tsa_deform_kernel(
    const float* __restrict__ v, const float* __restrict__ ref2d,
    const float* __restrict__ off, const float* __restrict__ aw,
    float* __restrict__ out)
{
  const int q = blockIdx.x;
  const int h = threadIdx.x >> 5;
  const int d = threadIdx.x & 31;
  const float refx = ref2d[q * 2 + 0];
  const float refy = ref2d[q * 2 + 1];
  float acc = 0.0f;
#pragma unroll
  for (int t = 0; t < 2; ++t) {
    const float* vt = v + (size_t)t * NQ * C_DIM;
#pragma unroll
    for (int p = 0; p < TPTS; ++p) {
      const int base = (h * 2 + t) * TPTS + p;
      const float ox = off[(size_t)q * 128 + base * 2 + 0];
      const float oy = off[(size_t)q * 128 + base * 2 + 1];
      const float wa = aw[(size_t)q * 64 + base];
      const float x = (refx + ox * 0.01f) * 100.0f - 0.5f;
      const float y = (refy + oy * 0.01f) * 100.0f - 0.5f;
      const float x0 = floorf(x), y0 = floorf(y);
#pragma unroll
      for (int dx = 0; dx < 2; ++dx)
#pragma unroll
        for (int dy = 0; dy < 2; ++dy) {
          const float xi = x0 + dx, yi = y0 + dy;
          if (xi >= 0.0f && xi <= 99.0f && yi >= 0.0f && yi <= 99.0f) {
            const float wgt = (1.0f - fabsf(x - xi)) * (1.0f - fabsf(y - yi));
            const int idx = (int)yi * 100 + (int)xi;
            acc += wa * wgt * vt[(size_t)idx * C_DIM + h * HDIM + d];
          }
        }
    }
  }
  out[(size_t)q * C_DIM + h * HDIM + d] = acc * 0.5f;
}

// ---------------------------------------------------------------------------
// MSDA (spatial cross-attn) deformable gather + cam-mask combine.
// ---------------------------------------------------------------------------
__global__ __launch_bounds__(256) void msda_deform_kernel(
    const float* __restrict__ v, const float* __restrict__ refcam,
    const float* __restrict__ off, const float* __restrict__ aw,
    const unsigned char* __restrict__ mask, float* __restrict__ slots)
{
  __shared__ float s_m[CAMS];
  __shared__ float s_cnt;
  const int q = blockIdx.x;
  const int h = threadIdx.x >> 5;
  const int d = threadIdx.x & 31;
  if (threadIdx.x < CAMS) {
    const unsigned char* mq = mask + ((size_t)threadIdx.x * NQ + q) * 4;
    s_m[threadIdx.x] = (mq[0] | mq[1] | mq[2] | mq[3]) ? 1.0f : 0.0f;
  }
  __syncthreads();
  if (threadIdx.x == 0) {
    float c = 0.0f;
    for (int i = 0; i < CAMS; ++i) c += s_m[i];
    s_cnt = c < 1.0f ? 1.0f : c;
  }
  __syncthreads();

  float acc = 0.0f;
  for (int cam = 0; cam < CAMS; ++cam) {
    if (s_m[cam] == 0.0f) continue;
    const float* vc = v + (size_t)cam * NUM_VAL * C_DIM;
    const float* rq = refcam + ((size_t)cam * NQ + q) * 8;
#pragma unroll
    for (int l = 0; l < LEVELS; ++l) {
      const int lw = kLvlW[l], lh = kLvlH[l], st = kLvlS[l];
      const float inw = 1.0f / (float)lw, inh = 1.0f / (float)lh;
#pragma unroll
      for (int p = 0; p < SPTS; ++p) {
        const int z  = p & 3;
        const int oi = ((h * LEVELS + l) * SPTS + p);
        const float lx = rq[z * 2 + 0] + off[(size_t)q * 512 + oi * 2 + 0] * inw;
        const float ly = rq[z * 2 + 1] + off[(size_t)q * 512 + oi * 2 + 1] * inh;
        const float wa = aw[(size_t)q * 256 + oi];
        const float x = lx * (float)lw - 0.5f;
        const float y = ly * (float)lh - 0.5f;
        const float x0 = floorf(x), y0 = floorf(y);
#pragma unroll
        for (int dx = 0; dx < 2; ++dx)
#pragma unroll
          for (int dy = 0; dy < 2; ++dy) {
            const float xi = x0 + dx, yi = y0 + dy;
            if (xi >= 0.0f && xi <= (float)(lw - 1) &&
                yi >= 0.0f && yi <= (float)(lh - 1)) {
              const float wgt = (1.0f - fabsf(x - xi)) * (1.0f - fabsf(y - yi));
              const int idx = st + (int)yi * lw + (int)xi;
              acc += wa * wgt * vc[(size_t)idx * C_DIM + h * HDIM + d];
            }
          }
      }
    }
  }
  slots[(size_t)q * C_DIM + h * HDIM + d] = acc / s_cnt;
}

// ---------------------------------------------------------------------------
extern "C" void kernel_launch(void* const* d_in, const int* in_sizes, int n_in,
                              void* d_out, int out_size, void* d_ws, size_t ws_size,
                              hipStream_t stream)
{
  const float* query    = (const float*)d_in[0];
  const float* value    = (const float*)d_in[2];   // (6,19560,1,256) == (117360,256)
  const float* bev_pos  = (const float*)d_in[3];
  const float* prev_bev = (const float*)d_in[4];
  const float* ref2d    = (const float*)d_in[5];
  const float* refcam   = (const float*)d_in[6];
  const unsigned char* mask = (const unsigned char*)d_in[7]; // bool bytes

  const float* tsa_off_w = (const float*)d_in[12];
  const float* tsa_off_b = (const float*)d_in[13];
  const float* tsa_aw_w  = (const float*)d_in[14];
  const float* tsa_aw_b  = (const float*)d_in[15];
  const float* tsa_v_w   = (const float*)d_in[16];
  const float* tsa_v_b   = (const float*)d_in[17];
  const float* tsa_o_w   = (const float*)d_in[18];
  const float* tsa_o_b   = (const float*)d_in[19];
  const float* ms_off_w  = (const float*)d_in[20];
  const float* ms_off_b  = (const float*)d_in[21];
  const float* ms_aw_w   = (const float*)d_in[22];
  const float* ms_aw_b   = (const float*)d_in[23];
  const float* ms_v_w    = (const float*)d_in[24];
  const float* ms_v_b    = (const float*)d_in[25];
  const float* sca_w     = (const float*)d_in[26];
  const float* sca_b     = (const float*)d_in[27];
  const float* n1g = (const float*)d_in[28];
  const float* n1b = (const float*)d_in[29];
  const float* n2g = (const float*)d_in[30];
  const float* n2b = (const float*)d_in[31];
  const float* n3g = (const float*)d_in[32];
  const float* n3b = (const float*)d_in[33];
  const float* ffn_w1 = (const float*)d_in[34];
  const float* ffn_b1 = (const float*)d_in[35];
  const float* ffn_w2 = (const float*)d_in[36];
  const float* ffn_b2 = (const float*)d_in[37];

  float* ws = (float*)d_ws;
  // workspace layout (floats); buffers reused where lifetimes allow
  float* vmsda    = ws;                                // 117360*256 = 30,044,160
  float* qc       = vmsda + (size_t)117360 * 256;      //  5,120,000 (later: msda_off)
  float* vtsa     = qc + (size_t)NQ * 512;             //  5,120,000 (later: ffn mid)
  float* tsa_off  = vtsa + (size_t)2 * NQ * C_DIM;     //  1,280,000
  float* tsa_aw   = tsa_off + (size_t)NQ * 128;        //    640,000
  float* tsa_out  = tsa_aw + (size_t)NQ * 64;          //  2,560,000 (later: msda_aw)
  float* tmp      = tsa_out + (size_t)NQ * C_DIM;      //  2,560,000
  float* q1       = tmp + (size_t)NQ * C_DIM;          //  2,560,000
  float* slotbuf  = q1 + (size_t)NQ * C_DIM;           //  2,560,000
  float* q2       = slotbuf + (size_t)NQ * C_DIM;      //  2,560,000
  float* msda_off = qc;
  float* msda_aw  = tsa_out;
  float* ffn_mid  = vtsa;

  // f16 weight pool (starts 16B-aligned: offset is a multiple of 4 floats)
  _Float16* hpool = (_Float16*)(q2 + (size_t)NQ * C_DIM);
  _Float16* wh_tsa_off = hpool;                        // 128*512
  _Float16* wh_tsa_aw  = wh_tsa_off + 128 * 512;       //  64*512
  _Float16* wh_tsa_v   = wh_tsa_aw  +  64 * 512;       // 256*256
  _Float16* wh_tsa_o   = wh_tsa_v   + 256 * 256;       // 256*256
  _Float16* wh_ms_off  = wh_tsa_o   + 256 * 256;       // 512*256
  _Float16* wh_ms_aw   = wh_ms_off  + 512 * 256;       // 256*256
  _Float16* wh_ms_v    = wh_ms_aw   + 256 * 256;       // 256*256
  _Float16* wh_sca     = wh_ms_v    + 256 * 256;       // 256*256
  _Float16* wh_ffn1    = wh_sca     + 256 * 256;       // 512*256
  _Float16* wh_ffn2    = wh_ffn1    + 512 * 256;       // 256*512

  auto cvt = [&](const float* s, _Float16* d, int n) {
    cvt_f16_kernel<<<(n + 255) / 256, 256, 0, stream>>>(s, d, n);
  };
  cvt(tsa_off_w, wh_tsa_off, 128 * 512);
  cvt(tsa_aw_w,  wh_tsa_aw,   64 * 512);
  cvt(tsa_v_w,   wh_tsa_v,   256 * 256);
  cvt(tsa_o_w,   wh_tsa_o,   256 * 256);
  cvt(ms_off_w,  wh_ms_off,  512 * 256);
  cvt(ms_aw_w,   wh_ms_aw,   256 * 256);
  cvt(ms_v_w,    wh_ms_v,    256 * 256);
  cvt(sca_w,     wh_sca,     256 * 256);
  cvt(ffn_w1,    wh_ffn1,    512 * 256);
  cvt(ffn_w2,    wh_ffn2,    256 * 512);

  auto gemm = [&](const float* A, const _Float16* Wh, const float* B, float* O,
                  int M, int N, int K, int act) {
    dim3 grid(N / 64, (M / 16 + 7) / 8);
    size_t smem = (size_t)64 * K * sizeof(_Float16);   // <= 64KB
    gemm_wmma_kernel<<<grid, 256, smem, stream>>>(A, Wh, B, O, M, N, K, act);
  };

  // ---- TSA -------------------------------------------------------------
  build_qc_kernel<<<(NQ * 512 + 255) / 256, 256, 0, stream>>>(query, bev_pos, qc);
  gemm(query,    wh_tsa_v, tsa_v_b, vtsa,                    NQ, 256, 256, 0);
  gemm(prev_bev, wh_tsa_v, tsa_v_b, vtsa + (size_t)NQ * 256, NQ, 256, 256, 0);
  gemm(qc, wh_tsa_off, tsa_off_b, tsa_off, NQ, 128, 512, 0);
  gemm(qc, wh_tsa_aw,  tsa_aw_b,  tsa_aw,  NQ,  64, 512, 0);
  softmax_groups_kernel<<<(NQ * 16 + 255) / 256, 256, 0, stream>>>(tsa_aw, NQ * 16, 4);
  tsa_deform_kernel<<<NQ, 256, 0, stream>>>(vtsa, ref2d, tsa_off, tsa_aw, tsa_out);
  gemm(tsa_out, wh_tsa_o, tsa_o_b, tmp, NQ, 256, 256, 0);
  add_ln_kernel<<<NQ, 256, 0, stream>>>(tmp, query, n1g, n1b, q1);

  // ---- SCA (MSDA) ------------------------------------------------------
  gemm(value, wh_ms_v, ms_v_b, vmsda, 117360, 256, 256, 0);
  gemm(q1, wh_ms_off, ms_off_b, msda_off, NQ, 512, 256, 0);
  gemm(q1, wh_ms_aw,  ms_aw_b,  msda_aw,  NQ, 256, 256, 0);
  softmax_groups_kernel<<<(NQ * 8 + 255) / 256, 256, 0, stream>>>(msda_aw, NQ * 8, 32);
  msda_deform_kernel<<<NQ, 256, 0, stream>>>(vmsda, refcam, msda_off, msda_aw, mask, slotbuf);
  gemm(slotbuf, wh_sca, sca_b, tmp, NQ, 256, 256, 0);
  add_ln_kernel<<<NQ, 256, 0, stream>>>(tmp, q1, n2g, n2b, q2);

  // ---- FFN -------------------------------------------------------------
  gemm(q2, wh_ffn1, ffn_b1, ffn_mid, NQ, 512, 256, 1);   // ReLU
  gemm(ffn_mid, wh_ffn2, ffn_b2, tmp, NQ, 256, 512, 0);
  add_ln_kernel<<<NQ, 256, 0, stream>>>(tmp, nullptr, n3g, n3b, (float*)d_out);
}
